// TimeDecoder_current_25503515803678
// MI455X (gfx1250) — matrix operand order repaired and checked by reading, not compiled, not run on hardware
//
#include <hip/hip_runtime.h>
#include <hip/hip_bf16.h>
#include <math.h>

// ---------------------------------------------------------------------------
// Types for CDNA5 WMMA
// ---------------------------------------------------------------------------
typedef __attribute__((ext_vector_type(16))) _Float16 v16h;
typedef __attribute__((ext_vector_type(4)))  _Float16 v4h;
typedef __attribute__((ext_vector_type(8)))  float    v8f;
typedef __attribute__((ext_vector_type(4)))  float    v4f;

#define ACT_NONE 0
#define ACT_GELU 1
#define ACT_SILU 2
#define ACT_SOFTPLUS 3
#define ACT_ELU1 4

__device__ __forceinline__ float act_apply(int act, float x) {
  switch (act) {
    case ACT_GELU:     return 0.5f * x * (1.f + erff(x * 0.7071067811865475f));
    case ACT_SILU:     return x / (1.f + __expf(-x));
    case ACT_SOFTPLUS: return (x > 20.f) ? x : log1pf(__expf(x));
    case ACT_ELU1:     return ((x > 0.f) ? x : (__expf(x) - 1.f)) + 1.0f + 1e-4f;
    default:           return x;
  }
}

// ---------------------------------------------------------------------------
// Generic WMMA GEMM:  C = act( A(MxK) * B + bias [+ resid] )
// B is (K,N) row-major if transB==0, else (N,K) row-major (i.e. W for x@W.T).
//
// Contract (all call sites obey): K, lda are multiples of 4 (ldb too except N=1,2) and every
// matrix is 16B aligned, so 4-wide chunks are either fully in-bounds in K
// or fully out (N may be ragged; handled per element on the notrans path).
//
// 64x64 block tile, 4 waves; wave w owns rows [w*16,w*16+16) x 64 cols =
// 4 WMMA tiles with the A fragment reused.
//
// LDS tiles are f16 in fragment-native order so each lane's 16-element
// fragment is one contiguous 32-byte LDS read (2x ds_load_b128):
//   A: stored [m][k] plain -> lane-half h reads K = h*16 .. h*16+15
//   B: stored [n][k] plain -> lane-half h reads K = h*16 .. h*16+15
// ---------------------------------------------------------------------------
__global__ __launch_bounds__(128)
void gemm_wmma_kernel(const float* __restrict__ A, const float* __restrict__ B,
                      const float* __restrict__ bias, const float* __restrict__ resid,
                      float* __restrict__ C,
                      int Mn, int Nn, int Kn, int lda, int ldb, int ldc,
                      int transB, int act)
{
  __shared__ __align__(64) _Float16 lA[64][40];  // [m][k], row stride 80B
  __shared__ __align__(64) _Float16 lB[64][40];  // [n][k]

  const int tid  = threadIdx.x;
  const int lane = tid & 31;
  const int wv   = tid >> 5;       // 0..3 : wave row strip
  const int m0   = blockIdx.y * 64;
  const int n0   = blockIdx.x * 64;
  const int half = lane >> 4;      // lane half (0/1)
  const int l16  = lane & 15;

  v8f acc[4] = {};

  for (int k0 = 0; k0 < Kn; k0 += 32) {
    // ---- Stage A tile (64 rows x 32 k): 4-wide vector chunks ----
    #pragma unroll
    for (int ch = 0; ch < 4; ch++) {
      int i  = ch * 128 + tid;        // chunk id 0..511
      int r  = i >> 3, c4 = (i & 7) << 2;
      int gm = m0 + r, gk = k0 + c4;
      v4f va = {};
      if (gm < Mn && gk < Kn) va = *(const v4f*)(A + (long long)gm * lda + gk);
      int pc = c4;
      v4h hv;
      hv.x = (_Float16)va.x; hv.y = (_Float16)va.y;
      hv.z = (_Float16)va.z; hv.w = (_Float16)va.w;
      *(v4h*)&lA[r][pc] = hv;         // one 8B LDS store
    }
    // ---- Stage B tile as [n][k] ----
    if (transB) {
      // B[n][k]: contiguous along k -> vector chunks like A
      #pragma unroll
      for (int ch = 0; ch < 4; ch++) {
        int i  = ch * 128 + tid;
        int nn = i >> 3, k4 = (i & 7) << 2;
        int gn = n0 + nn, gk = k0 + k4;
        v4f vb = {};
        if (gn < Nn && gk < Kn) vb = *(const v4f*)(B + (long long)gn * ldb + gk);
        v4h hv;
        hv.x = (_Float16)vb.x; hv.y = (_Float16)vb.y;
        hv.z = (_Float16)vb.z; hv.w = (_Float16)vb.w;
        *(v4h*)&lB[nn][k4] = hv;
      }
    } else {
      // B[k][n]: contiguous along n -> vector load, 4 scattered b16 stores
      #pragma unroll
      for (int ch = 0; ch < 4; ch++) {
        int i   = ch * 128 + tid;
        int kk  = i >> 4, n4 = (i & 15) << 2;
        int gk  = k0 + kk, gnb = n0 + n4;
        v4f vb = {};
        if (gk < Kn) {
          const float* bp = B + (long long)gk * ldb + gnb;
          if (gnb + 3 < Nn) {
            vb = *(const v4f*)bp;
          } else {                     // ragged N (predictor heads N=1,2)
            if (gnb + 0 < Nn) vb.x = bp[0];
            if (gnb + 1 < Nn) vb.y = bp[1];
            if (gnb + 2 < Nn) vb.z = bp[2];
            if (gnb + 3 < Nn) vb.w = bp[3];
          }
        }
        lB[n4 + 0][kk] = (_Float16)vb.x;
        lB[n4 + 1][kk] = (_Float16)vb.y;
        lB[n4 + 2][kk] = (_Float16)vb.z;
        lB[n4 + 3][kk] = (_Float16)vb.w;
      }
    }
    // Prefetch next A K-slab (speculative; dropped if OOB)
    if (k0 + 32 < Kn) {
      int pr = m0 + (tid >> 1);
      __builtin_prefetch(&A[(long long)pr * lda + k0 + 32 + ((tid & 1) << 4)], 0, 1);
    }
    __syncthreads();

    // One contiguous 32B fragment load per matrix per lane
    v16h af = *(const v16h*)&lA[wv * 16 + l16][half * 16];
    #pragma unroll
    for (int j = 0; j < 4; j++) {
      v16h bf = *(const v16h*)&lB[j * 16 + l16][half * 16];
      acc[j] = __builtin_amdgcn_wmma_f32_16x16x32_f16(
          false, af, false, bf, (short)0, acc[j], false, false);
    }
    __syncthreads();
  }

  // Epilogue: C/D layout -> row = r + 8*half, col = l16 (per 16x16 tile)
  const int rowBase = m0 + wv * 16 + half * 8;
  #pragma unroll
  for (int j = 0; j < 4; j++) {
    int col = n0 + j * 16 + l16;
    if (col < Nn) {
      float bv = bias ? bias[col] : 0.f;
      #pragma unroll
      for (int r = 0; r < 8; r++) {
        int row = rowBase + r;
        if (row < Mn) {
          float v = acc[j][r] + bv;
          if (resid) v += resid[(long long)row * ldc + col];
          C[(long long)row * ldc + col] = act_apply(act, v);
        }
      }
    }
  }
}

// ---------------------------------------------------------------------------
// LayerNorm over last dim (C=128). One wave per row, float4 I/O.
// ---------------------------------------------------------------------------
__global__ __launch_bounds__(128)
void layernorm_kernel(const float* __restrict__ x, const float* __restrict__ w,
                      const float* __restrict__ b, float* __restrict__ out, int rows)
{
  int row  = blockIdx.x * 4 + (threadIdx.x >> 5);
  int lane = threadIdx.x & 31;
  if (row >= rows) return;
  long long base = (long long)row * 128 + lane * 4;
  v4f v = *(const v4f*)(x + base);
  float s = v.x + v.y + v.z + v.w;
  #pragma unroll
  for (int o = 16; o > 0; o >>= 1) s += __shfl_xor(s, o, 32);
  float mu = s * (1.f / 128.f);
  float q = (v.x-mu)*(v.x-mu) + (v.y-mu)*(v.y-mu) + (v.z-mu)*(v.z-mu) + (v.w-mu)*(v.w-mu);
  #pragma unroll
  for (int o = 16; o > 0; o >>= 1) q += __shfl_xor(q, o, 32);
  float inv = rsqrtf(q * (1.f / 128.f) + 1e-5f);
  v4f wv4 = *(const v4f*)(w + lane * 4);
  v4f bv4 = *(const v4f*)(b + lane * 4);
  v4f ov;
  ov.x = (v.x - mu) * inv * wv4.x + bv4.x;
  ov.y = (v.y - mu) * inv * wv4.y + bv4.y;
  ov.z = (v.z - mu) * inv * wv4.z + bv4.z;
  ov.w = (v.w - mu) * inv * wv4.w + bv4.w;
  *(v4f*)(out + base) = ov;
}

// ---------------------------------------------------------------------------
// Fused (a [+ b]) -> optional sum_out, RMSNorm(sum)*w -> norm_out. C=128.
// ---------------------------------------------------------------------------
__global__ __launch_bounds__(128)
void addrms_kernel(const float* __restrict__ a, const float* __restrict__ b,
                   const float* __restrict__ w, float* __restrict__ sum_out,
                   float* __restrict__ norm_out, int rows)
{
  int row  = blockIdx.x * 4 + (threadIdx.x >> 5);
  int lane = threadIdx.x & 31;
  if (row >= rows) return;
  long long base = (long long)row * 128 + lane * 4;
  v4f v = *(const v4f*)(a + base);
  if (b) {
    v4f u = *(const v4f*)(b + base);
    v.x += u.x; v.y += u.y; v.z += u.z; v.w += u.w;
  }
  float s2 = v.x*v.x + v.y*v.y + v.z*v.z + v.w*v.w;
  #pragma unroll
  for (int o = 16; o > 0; o >>= 1) s2 += __shfl_xor(s2, o, 32);
  float inv = rsqrtf(s2 * (1.f / 128.f) + 1e-5f);
  if (sum_out) *(v4f*)(sum_out + base) = v;
  v4f wv4 = *(const v4f*)(w + lane * 4);
  v4f ov;
  ov.x = v.x * inv * wv4.x; ov.y = v.y * inv * wv4.y;
  ov.z = v.z * inv * wv4.z; ov.w = v.w * inv * wv4.w;
  *(v4f*)(norm_out + base) = ov;
}

// ---------------------------------------------------------------------------
// Fused attention, online softmax. Dh=16, H=8, C=128. One thread per (b,h,l).
// float4 vectorized q/k/v access.
// ---------------------------------------------------------------------------
__global__ __launch_bounds__(64)
void attention_kernel(const float* __restrict__ q, const float* __restrict__ k,
                      const float* __restrict__ v, const unsigned char* __restrict__ mask,
                      float* __restrict__ out, int Lq, int Lk)
{
  int l = blockIdx.x * 64 + threadIdx.x;
  int h = blockIdx.y, b = blockIdx.z;
  if (l >= Lq) return;
  const float* qp = q + ((long long)(b * Lq + l) * 128) + h * 16;
  v4f qv[4];
  #pragma unroll
  for (int j = 0; j < 4; j++) {
    qv[j] = *(const v4f*)(qp + 4 * j);
    qv[j].x *= 0.25f; qv[j].y *= 0.25f; qv[j].z *= 0.25f; qv[j].w *= 0.25f;
  }
  float m = -INFINITY, ssum = 0.f;
  v4f o[4] = {};
  for (int n = 0; n < Lk; n++) {
    const float* kp = k + ((long long)(b * Lk + n) * 128) + h * 16;
    float s = 0.f;
    #pragma unroll
    for (int j = 0; j < 4; j++) {
      v4f kv = *(const v4f*)(kp + 4 * j);
      s += qv[j].x * kv.x + qv[j].y * kv.y + qv[j].z * kv.z + qv[j].w * kv.w;
    }
    if (mask[b * Lk + n]) s = -1e9f;
    float mn = fmaxf(m, s);
    float corr = __expf(m - mn);
    float p = __expf(s - mn);
    ssum = ssum * corr + p;
    const float* vp = v + ((long long)(b * Lk + n) * 128) + h * 16;
    #pragma unroll
    for (int j = 0; j < 4; j++) {
      v4f vv = *(const v4f*)(vp + 4 * j);
      o[j].x = o[j].x * corr + p * vv.x;
      o[j].y = o[j].y * corr + p * vv.y;
      o[j].z = o[j].z * corr + p * vv.z;
      o[j].w = o[j].w * corr + p * vv.w;
    }
    m = mn;
  }
  float inv = 1.f / ssum;
  float* op = out + ((long long)(b * Lq + l) * 128) + h * 16;
  #pragma unroll
  for (int j = 0; j < 4; j++) {
    v4f ov; ov.x = o[j].x * inv; ov.y = o[j].y * inv;
    ov.z = o[j].z * inv; ov.w = o[j].w * inv;
    *(v4f*)(op + 4 * j) = ov;
  }
}

// ---------------------------------------------------------------------------
// Depthwise causal conv1d (K=4) + bias + SiLU. Reads xi (cols 0..255 of xz,
// row stride 512) with optional time reversal; writes xc in scan order.
// ---------------------------------------------------------------------------
__global__ __launch_bounds__(256)
void conv_silu_kernel(const float* __restrict__ xz, const float* __restrict__ w,
                      const float* __restrict__ bias, float* __restrict__ out, int rev)
{
  int idx = blockIdx.x * 256 + threadIdx.x;
  if (idx >= 96 * 60 * 256) return;
  int d = idx & 255;
  int t = (idx >> 8) % 60;
  int b = idx / (256 * 60);
  float acc = bias[d];
  #pragma unroll
  for (int j = 0; j < 4; j++) {
    int s = t - 3 + j;
    if (s >= 0) {
      int o = rev ? (59 - s) : s;
      acc += w[d * 4 + j] * xz[((long long)(b * 60 + o)) * 512 + d];
    }
  }
  out[(long long)idx] = acc / (1.f + __expf(-acc));
}

// ---------------------------------------------------------------------------
// Selective scan. One 16-lane group per (b,d) channel: lane = state index n,
// sequential over T=60 in scan order; y written back to original time index,
// gated by silu(z). Half-wave butterfly reduction over the 16 states.
// ---------------------------------------------------------------------------
__global__ __launch_bounds__(256)
void scan_kernel(const float* __restrict__ xc, const float* __restrict__ dt,
                 const float* __restrict__ xdb, const float* __restrict__ xz,
                 const float* __restrict__ A_log, const float* __restrict__ Dp,
                 float* __restrict__ ysum, int rev, int addout)
{
  int n   = threadIdx.x & 15;
  int grp = threadIdx.x >> 4;           // 0..15
  int p   = blockIdx.x * 16 + grp;      // (b,d) pair index
  if (p >= 96 * 256) return;
  int b = p >> 8, d = p & 255;
  float A  = -__expf(A_log[d * 16 + n]);
  float Dd = Dp[d];
  float h = 0.f;
  for (int t = 0; t < 60; t++) {
    long long rowi = (long long)(b * 60 + t);
    float u   = xc[rowi * 256 + d];
    float dtv = dt[rowi * 256 + d];
    float Bn  = xdb[rowi * 40 + 8 + n];
    float Cn  = xdb[rowi * 40 + 24 + n];
    h = __expf(dtv * A) * h + dtv * Bn * u;
    float yv = h * Cn;
    #pragma unroll
    for (int o = 8; o > 0; o >>= 1) yv += __shfl_xor(yv, o, 16);
    if (n == 0) {
      int to = rev ? (59 - t) : t;
      float z = xz[((long long)(b * 60 + to)) * 512 + 256 + d];
      float g = z / (1.f + __expf(-z));
      float val = (yv + u * Dd) * g;
      long long oi = ((long long)(b * 60 + to)) * 256 + d;
      if (addout) ysum[oi] += val; else ysum[oi] = val;
    }
  }
}

// ---------------------------------------------------------------------------
// Max over T axis: (96,60,128) -> (96,128)
// ---------------------------------------------------------------------------
__global__ __launch_bounds__(256)
void maxt_kernel(const float* __restrict__ x, float* __restrict__ out)
{
  int idx = blockIdx.x * 256 + threadIdx.x;
  if (idx >= 96 * 128) return;
  int c = idx & 127, bm = idx >> 7;
  float m = -INFINITY;
  for (int t = 0; t < 60; t++)
    m = fmaxf(m, x[((long long)(bm * 60 + t)) * 128 + c]);
  out[(long long)bm * 128 + c] = m;
}

// ---------------------------------------------------------------------------
// Host-side launch helper
// ---------------------------------------------------------------------------
static void gemm(hipStream_t st, const float* A, const float* B, const float* bias,
                 const float* resid, float* C, int M, int N, int K,
                 int lda, int ldb, int ldc, int transB, int act)
{
  dim3 g((N + 63) / 64, (M + 63) / 64, 1), blk(128);
  gemm_wmma_kernel<<<g, blk, 0, st>>>(A, B, bias, resid, C, M, N, K,
                                      lda, ldb, ldc, transB, act);
}

// Model dims
#define ROWS 5760      // 16*360 == 96*60
#define ENC_ROWS 8192  // 16*512

// Param index map (jax pytree: sorted dict keys at every level)
// d_in: 0=mode 1=encoding 2=mask, then params:
//  3..66   cross1 (4 x 16), 67..130 cross2
//  131..198 mamba1 (4 x 17), 199..266 mamba2
//  267 norm_f1_w, 268 norm_f2_w, 269..280 pred1, 281..292 pred2
#define X_BK 0
#define X_BO 1
#define X_BQ 2
#define X_BV 3
#define X_LN1B 4
#define X_LN1W 5
#define X_LN2B 6
#define X_LN2W 7
#define X_MLPB1 8
#define X_MLPB2 9
#define X_MLPW1 10
#define X_MLPW2 11
#define X_WK 12
#define X_WO 13
#define X_WQ 14
#define X_WV 15
// mamba dir (sorted): A_log, D, conv_b, conv_w, dt_proj_b, dt_proj_w, x_proj_w
#define M_ALOG 0
#define M_D 1
#define M_CONVB 2
#define M_CONVW 3
#define M_DTPB 4
#define M_DTPW 5
#define M_XPW 6
// mamba layer: bwd(0..6) fwd(7..13) in_proj_w(14) norm_w(15) out_proj_w(16)
// pred (sorted): c_b1 c_b2 c_w1 c_w2 g_b1 g_b2 g_w1 g_w2 s_b1 s_b2 s_w1 s_w2
#define P_CB1 0
#define P_CB2 1
#define P_CW1 2
#define P_CW2 3
#define P_GB1 4
#define P_GB2 5
#define P_GW1 6
#define P_GW2 7
#define P_SB1 8
#define P_SB2 9
#define P_SW1 10
#define P_SW2 11

extern "C" void kernel_launch(void* const* d_in, const int* in_sizes, int n_in,
                              void* d_out, int out_size, void* d_ws, size_t ws_size,
                              hipStream_t stream)
{
  (void)in_sizes; (void)n_in; (void)out_size; (void)ws_size;
  auto F = [&](int i) -> const float* { return (const float*)d_in[i]; };

  const float* enc = F(1);
  const unsigned char* mask = (const unsigned char*)d_in[2];
  float* out = (float*)d_out;

  // Workspace layout (floats)
  float* ws = (float*)d_ws;
  size_t o = 0;
  float* x_cur = ws + o; o += 737280;   // current mode, (5760,128)
  float* xn    = ws + o; o += 737280;
  float* qb    = ws + o; o += 737280;
  float* kb    = ws + o; o += 1048576;  // (8192,128)
  float* vb    = ws + o; o += 1048576;
  float* ao    = ws + o; o += 737280;
  float* big   = ws + o; o += 2949120;  // MLP hidden (5760,512) / xz (5760,512)
  float* resid = ws + o; o += 737280;
  float* xcf   = ws + o; o += 1474560;  // (5760,256)
  float* xcb   = ws + o; o += 1474560;
  float* xdbf  = ws + o; o += 230400;   // (5760,40)
  float* xdbb  = ws + o; o += 230400;
  float* dtf   = ws + o; o += 1474560;
  float* dtb   = ws + o; o += 1474560;
  float* ysum  = ws + o; o += 1474560;
  float* tmpa  = ws + o; o += 368640;   // (5760,64)
  float* tmpb  = ws + o; o += 368640;
  float* xmb   = ws + o; o += 12288;    // (96,128)
  float* sb    = ws + o; o += 6144;     // (96,64)

  // mode -> working copy (never mutate inputs)
  hipMemcpyAsync(x_cur, d_in[0], (size_t)737280 * 4, hipMemcpyDeviceToDevice, stream);

  const dim3 rowsGrid(ROWS / 4), rowsBlk(128);

  for (int stage = 0; stage < 2; stage++) {
    int crossBase = stage == 0 ? 3 : 67;
    int mambaBase = stage == 0 ? 131 : 199;
    int normFIdx  = stage == 0 ? 267 : 268;

    for (int layer = 0; layer < 4; layer++) {
      int cb = crossBase + layer * 16;
      // ---- cross block (x: (16,360,128) flat (5760,128)) ----
      layernorm_kernel<<<rowsGrid, rowsBlk, 0, stream>>>(x_cur, F(cb + X_LN1W), F(cb + X_LN1B), xn, ROWS);
      gemm(stream, xn,  F(cb + X_WQ), F(cb + X_BQ), nullptr, qb, ROWS, 128, 128, 128, 128, 128, 0, ACT_NONE);
      gemm(stream, enc, F(cb + X_WK), F(cb + X_BK), nullptr, kb, ENC_ROWS, 128, 128, 128, 128, 128, 0, ACT_NONE);
      gemm(stream, enc, F(cb + X_WV), F(cb + X_BV), nullptr, vb, ENC_ROWS, 128, 128, 128, 128, 128, 0, ACT_NONE);
      attention_kernel<<<dim3(6, 8, 16), 64, 0, stream>>>(qb, kb, vb, mask, ao, 360, 512);
      gemm(stream, ao, F(cb + X_WO), F(cb + X_BO), x_cur, x_cur, ROWS, 128, 128, 128, 128, 128, 0, ACT_NONE);
      layernorm_kernel<<<rowsGrid, rowsBlk, 0, stream>>>(x_cur, F(cb + X_LN2W), F(cb + X_LN2B), xn, ROWS);
      gemm(stream, xn,  F(cb + X_MLPW1), F(cb + X_MLPB1), nullptr, big, ROWS, 512, 128, 128, 512, 512, 0, ACT_GELU);
      gemm(stream, big, F(cb + X_MLPW2), F(cb + X_MLPB2), x_cur, x_cur, ROWS, 128, 512, 512, 128, 128, 0, ACT_NONE);

      // ---- mamba block (x: (96,60,128)) ----
      int mb = mambaBase + layer * 17;
      int fw = mb + 7, bw = mb;  // fwd / bwd direction param bases
      addrms_kernel<<<rowsGrid, rowsBlk, 0, stream>>>(
          x_cur, layer == 0 ? nullptr : resid, F(mb + 15), resid, xn, ROWS);
      // xz = xn @ in_proj_w.T  (512,128)
      gemm(stream, xn, F(mb + 14), nullptr, nullptr, big, ROWS, 512, 128, 128, 128, 512, 1, ACT_NONE);
      // depthwise conv + silu (fwd / bwd in scan order)
      conv_silu_kernel<<<dim3(5760), 256, 0, stream>>>(big, F(fw + M_CONVW), F(fw + M_CONVB), xcf, 0);
      conv_silu_kernel<<<dim3(5760), 256, 0, stream>>>(big, F(bw + M_CONVW), F(bw + M_CONVB), xcb, 1);
      // xdb = xc @ x_proj_w.T  (40,256)
      gemm(stream, xcf, F(fw + M_XPW), nullptr, nullptr, xdbf, ROWS, 40, 256, 256, 256, 40, 1, ACT_NONE);
      gemm(stream, xcb, F(bw + M_XPW), nullptr, nullptr, xdbb, ROWS, 40, 256, 256, 256, 40, 1, ACT_NONE);
      // dt = softplus(dtr @ dt_proj_w.T + b)  (256,8); dtr = xdb[:, :8]
      gemm(stream, xdbf, F(fw + M_DTPW), F(fw + M_DTPB), nullptr, dtf, ROWS, 256, 8, 40, 8, 256, 1, ACT_SOFTPLUS);
      gemm(stream, xdbb, F(bw + M_DTPW), F(bw + M_DTPB), nullptr, dtb, ROWS, 256, 8, 40, 8, 256, 1, ACT_SOFTPLUS);
      // selective scan, gated by silu(z); bwd accumulates into ysum
      scan_kernel<<<dim3(1536), 256, 0, stream>>>(xcf, dtf, xdbf, big, F(fw + M_ALOG), F(fw + M_D), ysum, 0, 0);
      scan_kernel<<<dim3(1536), 256, 0, stream>>>(xcb, dtb, xdbb, big, F(bw + M_ALOG), F(bw + M_D), ysum, 1, 1);
      // out = ysum @ out_proj_w.T  (128,256)
      gemm(stream, ysum, F(mb + 16), nullptr, nullptr, x_cur, ROWS, 128, 256, 256, 256, 128, 1, ACT_NONE);
    }
    // final fused add + RMSNorm (in place)
    addrms_kernel<<<rowsGrid, rowsBlk, 0, stream>>>(x_cur, resid, F(normFIdx), nullptr, x_cur, ROWS);

    // ---- predictor ----
    int pb = stage == 0 ? 269 : 281;
    float* out_res  = out + (stage == 0 ? 0      : 748896);
    float* out_pi   = out + (stage == 0 ? 11520  : 760416);
    float* out_scal = out + (stage == 0 ? 760512 : 772032);
    gemm(stream, x_cur, F(pb + P_GW1), F(pb + P_GB1), nullptr, tmpa, ROWS, 64, 128, 128, 64, 64, 0, ACT_GELU);
    gemm(stream, tmpa,  F(pb + P_GW2), F(pb + P_GB2), nullptr, out_res, ROWS, 2, 64, 64, 2, 2, 0, ACT_NONE);
    gemm(stream, x_cur, F(pb + P_CW1), F(pb + P_CB1), nullptr, tmpb, ROWS, 64, 128, 128, 64, 64, 0, ACT_GELU);
    gemm(stream, tmpb,  F(pb + P_CW2), F(pb + P_CB2), nullptr, out_scal, ROWS, 2, 64, 64, 2, 2, 0, ACT_ELU1);
    maxt_kernel<<<dim3(48), 256, 0, stream>>>(x_cur, xmb);
    gemm(stream, xmb, F(pb + P_SW1), F(pb + P_SB1), nullptr, sb, 96, 64, 128, 128, 64, 64, 0, ACT_GELU);
    gemm(stream, sb,  F(pb + P_SW2), F(pb + P_SB2), nullptr, out_pi, 96, 1, 64, 64, 1, 1, 0, ACT_NONE);
  }

  // returned mode = post-stage-2 mode
  hipMemcpyAsync(out + 11616, x_cur, (size_t)737280 * 4, hipMemcpyDeviceToDevice, stream);
}
